// CrossAttention2_67465346286179
// MI455X (gfx1250) — matrix operand order repaired and checked by reading, not compiled, not run on hardware
//
#include <hip/hip_runtime.h>
#include <hip/hip_bf16.h>
#include <math.h>

// Problem constants (match reference)
#define BB 8
#define NN 256
#define MM 1024
#define CC 1024
#define HH 16
#define DD 64
#define LN_EPS 1e-5f

typedef __attribute__((ext_vector_type(16))) _Float16 v16h;
typedef __attribute__((ext_vector_type(8)))  _Float16 v8h;
typedef __attribute__((ext_vector_type(8)))  float    v8f;

#define WMMA_F16(a, b, c) \
  __builtin_amdgcn_wmma_f32_16x16x32_f16(false, (a), false, (b), (short)0, (c), false, false)

// ---- CDNA5 async global->LDS copy (ASYNCcnt) with sync fallback -----------
#if defined(__gfx1250__) && \
    __has_builtin(__builtin_amdgcn_global_load_async_to_lds_b128) && \
    __has_builtin(__builtin_amdgcn_s_wait_asynccnt)
#define USE_ASYNC_CP 1
#else
#define USE_ASYNC_CP 0
#endif

#if USE_ASYNC_CP
typedef int v4i __attribute__((ext_vector_type(4)));
typedef __attribute__((address_space(1))) v4i v4i_glb;  // global (AS1) int4
typedef __attribute__((address_space(3))) v4i v4i_lds;  // LDS (AS3) int4
// Generic shared-memory addresses carry the LDS byte offset in the low 32 bits
// (ISA 10.2); AS1 generic global addresses are identity.
static __device__ __forceinline__ v4i_glb* to_glb_v4i(const void* p) {
  return (v4i_glb*)(unsigned long long)p;
}
static __device__ __forceinline__ v4i_lds* to_lds_v4i(void* p) {
  return (v4i_lds*)(unsigned int)(unsigned long long)p;
}
#define ASYNC_CP_B128(gsrc, ldst) \
  __builtin_amdgcn_global_load_async_to_lds_b128(to_glb_v4i(gsrc), to_lds_v4i(ldst), 0, 0)
#define WAIT_ASYNC(n) __builtin_amdgcn_s_wait_asynccnt(n)
#else
#define WAIT_ASYNC(n)
#endif

// A fragment 16x32 f16 (row = M = lane%16) from a K-contiguous f16 row.
// A layout (matches B): lanes 0-15 hold K 0..15; lanes 16-31 hold K 16..31.
// Two aligned 16-byte loads, no conversion.
static __device__ __forceinline__ v16h load_a_frag_f16(const _Float16* __restrict__ row,
                                                       int lane) {
  const int aoff = (lane < 16) ? 0 : 16;
  const v8h lo = *(const v8h*)(row + aoff);
  const v8h hi = *(const v8h*)(row + aoff + 8);
  v16h a;
#pragma unroll
  for (int j = 0; j < 8; ++j) { a[j] = lo[j]; a[j + 8] = hi[j]; }
  return a;
}

// A fragment from fp32 data (used for the LDS probability tile in attention).
static __device__ __forceinline__ v16h load_a_frag_f32(const float* __restrict__ row,
                                                       int k0, int lane) {
  const int aoff = (lane < 16) ? 0 : 16;
  v16h a;
#pragma unroll
  for (int j = 0; j < 8; ++j) {
    a[j]     = (_Float16)row[k0 + aoff + j];
    a[j + 8] = (_Float16)row[k0 + aoff + 8 + j];
  }
  return a;
}

// B fragment 32x16 f16: lane = column N, lanes 0-15 carry K=0..15,
// lanes 16-31 carry K=16..31. K-contiguous source: one 32-byte vector load.
static __device__ __forceinline__ v16h load_b_frag_f16(const _Float16* __restrict__ col,
                                                       int boff) {
  return *(const v16h*)(col + boff);
}

// ---------------------------------------------------------------------------
// Streaming fp32 -> f16 conversion (8 elements/thread). n % 8 == 0.
// ---------------------------------------------------------------------------
__global__ __launch_bounds__(256) void cvt_f32_to_f16(const float* __restrict__ s,
                                                      _Float16* __restrict__ d, int n) {
  const int i = (blockIdx.x * blockDim.x + threadIdx.x) * 8;
  if (i + 8 > n) return;
  const float4 a = *(const float4*)(s + i);
  const float4 b = *(const float4*)(s + i + 4);
  v8h o;
  o[0] = (_Float16)a.x; o[1] = (_Float16)a.y; o[2] = (_Float16)a.z; o[3] = (_Float16)a.w;
  o[4] = (_Float16)b.x; o[5] = (_Float16)b.y; o[6] = (_Float16)b.z; o[7] = (_Float16)b.w;
  *(v8h*)(d + i) = o;
}

// ---------------------------------------------------------------------------
// GEMM: out = X @ W^T (+bias), X/W f16 K-major. LDS-tiled, double-buffered,
// async-staged. Block = 256 threads (8 waves) -> 128x64 output tile; each
// wave owns 16 rows with 4 accumulators (16x64), so one A fragment feeds
// 4 back-to-back v_wmma_f32_16x16x32_f16 fed by pure ds_load_b128 (no cvt).
// OUT_F32: write fp32 (+bias); else write f16 (bias ignored).
// Requires rows%128==0, cols%64==0, kdim%32==0 (true for all launches here).
// ---------------------------------------------------------------------------
template <bool OUT_F32>
__global__ __launch_bounds__(256) void gemm_nt_wmma_f16(
    const _Float16* __restrict__ X, const _Float16* __restrict__ W,
    const float* __restrict__ bias, void* __restrict__ outv,
    int cols, int kdim) {
  __shared__ __align__(32) _Float16 Atile[2][128 * 32];  // 2 x 8 KB
  __shared__ __align__(32) _Float16 Btile[2][64 * 32];   // 2 x 4 KB

  const int t    = threadIdx.x;
  const int lane = t & 31;
  const int wave = t >> 5;
  const int row0 = blockIdx.y * 128;
  const int col0 = blockIdx.x * 64;
  const int rA   = lane & 15;
  const int boff = (lane < 16) ? 0 : 16;
  const int mb   = (lane < 16) ? 0 : 8;
  const int nk   = kdim >> 5;

  // Stage one 128x32 A tile + 64x32 B tile (f16) into LDS buffer bi.
  // 3 x b128 per thread -> ASYNCcnt depth 3 per stage.
  auto stage = [&](int bi, int k0) {
#pragma unroll
    for (int i = 0; i < 2; ++i) {                    // A: 512 16B chunks
      const int c = t + 256 * i;
      const int r = c >> 2, hc = (c & 3) << 3;       // 4 chunks (8 halfs each) per row
      const _Float16* g = X + (size_t)(row0 + r) * kdim + k0 + hc;
      _Float16* l = &Atile[bi][r * 32 + hc];
#if USE_ASYNC_CP
      ASYNC_CP_B128(g, l);
#else
      *(v8h*)l = *(const v8h*)g;
#endif
    }
    {                                                // B: 256 16B chunks
      const int c = t;
      const int r = c >> 2, hc = (c & 3) << 3;
      const _Float16* g = W + (size_t)(col0 + r) * kdim + k0 + hc;
      _Float16* l = &Btile[bi][r * 32 + hc];
#if USE_ASYNC_CP
      ASYNC_CP_B128(g, l);
#else
      *(v8h*)l = *(const v8h*)g;
#endif
    }
  };

  v8f acc[4] = {v8f{}, v8f{}, v8f{}, v8f{}};

  auto compute = [&](int bi) {
    const v16h af = load_a_frag_f16(&Atile[bi][(wave * 16 + rA) * 32], lane);
    v16h bf[4];
#pragma unroll
    for (int nt2 = 0; nt2 < 4; ++nt2)
      bf[nt2] = load_b_frag_f16(&Btile[bi][(nt2 * 16 + rA) * 32], boff);
#pragma unroll
    for (int nt2 = 0; nt2 < 4; ++nt2)
      acc[nt2] = WMMA_F16(af, bf[nt2], acc[nt2]);
  };

  stage(0, 0);
  int ti = 0;
  for (; ti + 1 < nk; ++ti) {
    stage((ti + 1) & 1, (ti + 1) << 5);  // prefetch next K-slab
    WAIT_ASYNC(3);                       // oldest stage's 3 copies complete
    __syncthreads();
    compute(ti & 1);
    __syncthreads();                     // protect buffer before re-stage
  }
  WAIT_ASYNC(0);
  __syncthreads();
  compute(ti & 1);

#pragma unroll
  for (int nt2 = 0; nt2 < 4; ++nt2) {
    const int n = col0 + nt2 * 16 + rA;
    if constexpr (OUT_F32) {
      float* out = (float*)outv;
      const float bv = bias ? bias[n] : 0.0f;
#pragma unroll
      for (int v = 0; v < 8; ++v)
        out[(size_t)(row0 + wave * 16 + mb + v) * cols + n] = acc[nt2][v] + bv;
    } else {
      _Float16* out = (_Float16*)outv;
#pragma unroll
      for (int v = 0; v < 8; ++v)
        out[(size_t)(row0 + wave * 16 + mb + v) * cols + n] = (_Float16)acc[nt2][v];
    }
  }
}

// ---------------------------------------------------------------------------
// Per-head LayerNorm over D=64 contiguous f16 values (fp32 math). One wave32
// per row.
// ---------------------------------------------------------------------------
__global__ __launch_bounds__(256) void ln_heads_f16(_Float16* __restrict__ x,
                                                    const float* __restrict__ gamma,
                                                    const float* __restrict__ beta,
                                                    int nrows) {
  const int lane = threadIdx.x & 31;
  const int wave = threadIdx.x >> 5;
  const int row  = blockIdx.x * (blockDim.x >> 5) + wave;
  if (row >= nrows) return;
  _Float16* p = x + (size_t)row * DD;
  float v0 = (float)p[lane], v1 = (float)p[lane + 32];
  float s  = v0 + v1;
  float s2 = v0 * v0 + v1 * v1;
#pragma unroll
  for (int off = 16; off; off >>= 1) {
    s  += __shfl_xor(s, off, 32);
    s2 += __shfl_xor(s2, off, 32);
  }
  const float mu  = s * (1.0f / DD);
  const float var = s2 * (1.0f / DD) - mu * mu;
  const float rs  = rsqrtf(var + LN_EPS);
  p[lane]      = (_Float16)((v0 - mu) * rs * gamma[lane]      + beta[lane]);
  p[lane + 32] = (_Float16)((v1 - mu) * rs * gamma[lane + 32] + beta[lane + 32]);
}

// ---------------------------------------------------------------------------
// Fused attention per (b, h, 16-query-row tile). Q/K/V are f16 [.,.,C].
//   phase 1: scores[16,1024] = (Q_tile @ K^T) * scale  -> LDS (WMMA, f16 frags)
//   phase 2: masked softmax in LDS; stream attn tile to d_out (non-temporal)
//   phase 3: out_tile[16,64] = P @ V (WMMA) -> f16 workspace [B,N,C]
// ---------------------------------------------------------------------------
__global__ __launch_bounds__(256) void attn_fused(
    const _Float16* __restrict__ Q,   // [B,N,C] post-LN
    const _Float16* __restrict__ Kc,  // [B,M,C] post-LN
    const _Float16* __restrict__ V,   // [B,M,C]
    const unsigned char* __restrict__ mask,  // [B,M] bool
    _Float16* __restrict__ attn_ws,   // [B,N,C] f16 (head-merged attn@V)
    float* __restrict__ attn_out)     // [B,H,N,M] -> d_out attn region
{
  __shared__ float sc[16 * MM];  // 64 KB score/prob tile

  const int nt   = blockIdx.x;          // query tile 0..15
  const int h    = blockIdx.y;          // head
  const int b    = blockIdx.z;          // batch
  const int lane = threadIdx.x & 31;
  const int wave = threadIdx.x >> 5;
  const int n0   = nt * 16;
  const int rA   = lane & 15;
  const int boff = (lane < 16) ? 0 : 16;
  const int mb   = (lane < 16) ? 0 : 8;
  const float scale = 0.125f;  // D^-0.5 = 64^-0.5

  const _Float16* qb = Q  + ((size_t)b * NN + n0) * CC + h * DD;
  const _Float16* kb = Kc + (size_t)b * MM * CC + h * DD;
  const _Float16* vb = V  + (size_t)b * MM * CC + h * DD;

  // ---- phase 1: scores -> LDS --------------------------------------------
  // Q fragments are invariant across m-tiles: hoist both K-halves.
  const _Float16* qrow = qb + (size_t)rA * CC;
  const v16h afq0 = load_a_frag_f16(qrow, lane);
  const v16h afq1 = load_a_frag_f16(qrow + 32, lane);
  for (int mt = wave; mt < MM / 16; mt += 8) {
    const _Float16* krow = kb + (size_t)(mt * 16 + rA) * CC;  // column m, contiguous in d
    const v16h bf0 = load_b_frag_f16(krow, boff);
    const v16h bf1 = load_b_frag_f16(krow + 32, boff);
    v8f acc = {};
    acc = WMMA_F16(afq0, bf0, acc);
    acc = WMMA_F16(afq1, bf1, acc);
    const int col = mt * 16 + rA;
#pragma unroll
    for (int v = 0; v < 8; ++v) sc[(mb + v) * MM + col] = acc[v] * scale;
  }
  __syncthreads();

  // ---- phase 2: masked softmax + stream attn (non-temporal) ---------------
  {
    const int r = threadIdx.x >> 4;  // 0..15; its 16 lanes sit in one wave-half
    const int c = threadIdx.x & 15;
    const unsigned char* mrow = mask + (size_t)b * MM;
    float mx = -INFINITY;
    for (int m = c; m < MM; m += 16) {
      float s = mrow[m] ? sc[r * MM + m] : -INFINITY;
      sc[r * MM + m] = s;
      mx = fmaxf(mx, s);
    }
#pragma unroll
    for (int off = 8; off; off >>= 1) mx = fmaxf(mx, __shfl_xor(mx, off, 32));
    float sum = 0.0f;
    for (int m = c; m < MM; m += 16) {
      float s = sc[r * MM + m];
      float e = (s == -INFINITY) ? 0.0f : __expf(s - mx);
      sc[r * MM + m] = e;
      sum += e;
    }
#pragma unroll
    for (int off = 8; off; off >>= 1) sum += __shfl_xor(sum, off, 32);
    const float inv = 1.0f / sum;
    float* arow = attn_out + (((size_t)b * HH + h) * NN + n0 + r) * MM;
    for (int m = c; m < MM; m += 16) {
      float p = sc[r * MM + m] * inv;
      sc[r * MM + m] = p;
      __builtin_nontemporal_store(p, &arow[m]);  // 134 MB stream: don't pollute L2
    }
  }
  __syncthreads();

  // ---- phase 3: out_tile = P @ V (waves 0..3 each take one 16-wide d tile) -
  if (wave < 4) {
    const int d0 = wave * 16;
    v8f acc = {};
    const float* prow = &sc[(size_t)rA * MM];
    for (int k0 = 0; k0 < MM; k0 += 64) {  // two K-steps per iteration
      const v16h af0 = load_a_frag_f32(prow, k0, lane);
      const v16h af1 = load_a_frag_f32(prow, k0 + 32, lane);
      v16h bf0, bf1;
#pragma unroll
      for (int j = 0; j < 16; ++j) {
        bf0[j] = vb[(size_t)(k0 + boff + j) * CC + d0 + rA];
        bf1[j] = vb[(size_t)(k0 + 32 + boff + j) * CC + d0 + rA];
      }
      acc = WMMA_F16(af0, bf0, acc);
      acc = WMMA_F16(af1, bf1, acc);
    }
#pragma unroll
    for (int v = 0; v < 8; ++v)
      attn_ws[((size_t)b * NN + n0 + mb + v) * CC + h * DD + d0 + rA] = (_Float16)acc[v];
  }
}

// ---------------------------------------------------------------------------
extern "C" void kernel_launch(void* const* d_in, const int* in_sizes, int n_in,
                              void* d_out, int out_size, void* d_ws, size_t ws_size,
                              hipStream_t stream) {
  (void)in_sizes; (void)n_in; (void)out_size; (void)ws_size;

  const float*         query   = (const float*)d_in[0];   // [B,N,C]
  const float*         context = (const float*)d_in[1];   // [B,M,C]
  const unsigned char* mask    = (const unsigned char*)d_in[2];  // [B,M] bool
  const float*         Wq      = (const float*)d_in[3];   // [C,C]
  const float*         Wk      = (const float*)d_in[4];
  const float*         Wv      = (const float*)d_in[5];
  const float*         q_gamma = (const float*)d_in[6];
  const float*         q_beta  = (const float*)d_in[7];
  const float*         k_gamma = (const float*)d_in[8];
  const float*         k_beta  = (const float*)d_in[9];
  const float*         Wp      = (const float*)d_in[10];
  const float*         bp      = (const float*)d_in[11];

  const size_t nQ = (size_t)BB * NN * CC;  // 2,097,152
  const size_t nK = (size_t)BB * MM * CC;  // 8,388,608
  const size_t nW = (size_t)CC * CC;       // 1,048,576

  // Workspace layout (all f16): inputs, weights, Q/K/V, attn@V
  _Float16* Xq16 = (_Float16*)d_ws;
  _Float16* Xc16 = Xq16 + nQ;
  _Float16* Wq16 = Xc16 + nK;
  _Float16* Wk16 = Wq16 + nW;
  _Float16* Wv16 = Wk16 + nW;
  _Float16* Wp16 = Wv16 + nW;
  _Float16* Qh   = Wp16 + nW;
  _Float16* Kh   = Qh + nQ;
  _Float16* Vh   = Kh + nK;
  _Float16* AOh  = Vh + nK;

  // Output tuple: out [B,N,C] fp32, then attn [B,H,N,M] fp32
  float* out_final = (float*)d_out;
  float* attn_out  = out_final + nQ;

  // f32 -> f16 streaming conversions (8 elems/thread)
  cvt_f32_to_f16<<<nQ / 2048, 256, 0, stream>>>(query,   Xq16, (int)nQ);
  cvt_f32_to_f16<<<nK / 2048, 256, 0, stream>>>(context, Xc16, (int)nK);
  cvt_f32_to_f16<<<nW / 2048, 256, 0, stream>>>(Wq, Wq16, (int)nW);
  cvt_f32_to_f16<<<nW / 2048, 256, 0, stream>>>(Wk, Wk16, (int)nW);
  cvt_f32_to_f16<<<nW / 2048, 256, 0, stream>>>(Wv, Wv16, (int)nW);
  cvt_f32_to_f16<<<nW / 2048, 256, 0, stream>>>(Wp, Wp16, (int)nW);

  const dim3 gridQ(CC / 64, (BB * NN) / 128);  // (16,16)
  const dim3 gridK(CC / 64, (BB * MM) / 128);  // (16,64)

  // QKV projections (X @ W^T), f16 in / f16 out
  gemm_nt_wmma_f16<false><<<gridQ, 256, 0, stream>>>(Xq16, Wq16, nullptr, Qh, CC, CC);
  gemm_nt_wmma_f16<false><<<gridK, 256, 0, stream>>>(Xc16, Wk16, nullptr, Kh, CC, CC);
  gemm_nt_wmma_f16<false><<<gridK, 256, 0, stream>>>(Xc16, Wv16, nullptr, Vh, CC, CC);

  // Per-head LayerNorm on q and k (rows of D=64, contiguous)
  ln_heads_f16<<<(BB * NN * HH) / 8, 256, 0, stream>>>(Qh, q_gamma, q_beta, BB * NN * HH);
  ln_heads_f16<<<(BB * MM * HH) / 8, 256, 0, stream>>>(Kh, k_gamma, k_beta, BB * MM * HH);

  // Fused scores + softmax + attn@V
  attn_fused<<<dim3(NN / 16, HH, BB), 256, 0, stream>>>(Qh, Kh, Vh, mask, AOh, attn_out);

  // Output projection: out = AO @ Wp^T + bp (f16 in, fp32 out)
  gemm_nt_wmma_f16<true><<<gridQ, 256, 0, stream>>>(AOh, Wp16, bp, out_final, CC, CC);
}
